// SvdWaveAttention_4045859193023
// MI455X (gfx1250) — matrix operand order repaired and checked by reading, not compiled, not run on hardware
//
#include <hip/hip_runtime.h>
#include <hip/hip_bf16.h>

// ---------------------------------------------------------------------------
// SvdWaveAttention for MI455X (gfx1250, wave32).
// Memory-bound (~400MB, ~1.5 GFLOP => ~17us at 23.3TB/s). Big passes use
// global_load_async_to_lds_b128 (ASYNCcnt); GEMMs use v_wmma_f32_16x16x32_bf16
// with fragment-major LDS staging (b128-only LDS traffic), K fully unrolled.
// ---------------------------------------------------------------------------

typedef __attribute__((ext_vector_type(16))) __bf16 v16bf;
typedef __attribute__((ext_vector_type(8)))  float  v8f;

static __device__ __forceinline__ unsigned short f2bf(float f) {
  unsigned int u = __builtin_bit_cast(unsigned int, f);
  unsigned int r = u + 0x7FFFu + ((u >> 16) & 1u);   // RNE
  return (unsigned short)(r >> 16);
}

// LDS offset of a __shared__ object (flat addr low 32 bits == LDS offset).
static __device__ __forceinline__ unsigned lds_off(const void* p) {
  return (unsigned)(unsigned long long)p;
}
// CDNA5 async global->LDS copy, 16 bytes per lane (tracked by ASYNCcnt).
static __device__ __forceinline__ void async_copy_b128(unsigned lds, const void* g) {
  asm volatile("global_load_async_to_lds_b128 %0, %1, off"
               :: "v"(lds), "v"(g) : "memory");
}
static __device__ __forceinline__ void wait_async() {
  asm volatile("s_wait_asynccnt 0" ::: "memory");
}

// ---------------------------------------------- cast + transpose f32[K,N]->bf16[N,K]
__global__ void k_cast_transpose(const float* __restrict__ src,
                                 unsigned short* __restrict__ dst, int K, int N) {
  int i = blockIdx.x * 256 + threadIdx.x;
  if (i < K * N) {
    int k = i / N, n = i % N;
    dst[(size_t)n * K + k] = f2bf(src[i]);
  }
}

// ------------------------------------------------- axis means, stage 1 (per b,c)
__global__ void k_means_partial(const float* __restrict__ x,
                                float* __restrict__ part) {
  __shared__ float A3[4096];
  __shared__ float A2[4096];
  __shared__ float A1[4096];
  int t = threadIdx.x;                     // 256
  for (int k = t; k < 4096; k += 256) { A3[k] = 0.f; A2[k] = 0.f; A1[k] = 0.f; }
  __syncthreads();
  int bc = blockIdx.x;                     // b*64 + c
  const float* base = x + (size_t)bc * 64 * 64 * 64;
  for (int n = 0; n < 64; ++n) {
    const float4* tp = (const float4*)(base + (size_t)n * 4096);
#pragma unroll
    for (int k = 0; k < 4; ++k) {
      int vec = t + k * 256;               // 0..1023
      float4 v = tp[vec];
      int e0 = vec * 4;
#pragma unroll
      for (int e = 0; e < 4; ++e) {
        int idx = e0 + e;
        int i = idx >> 6, j = idx & 63;
        float val = (&v.x)[e];
        A3[idx] += val;                          // owner-stable, race-free
        atomicAdd(&A2[(n << 6) + j], val);       // ds_add_f32
        atomicAdd(&A1[(i << 6) + n], val);
      }
    }
  }
  __syncthreads();
  float* out = part + (size_t)bc * 12288;
  for (int k = t; k < 4096; k += 256) {
    out[k] = A3[k]; out[4096 + k] = A2[k]; out[8192 + k] = A1[k];
  }
}

// ------------------------------------------------- axis means, stage 2
__global__ void k_means_final(const float* __restrict__ part,
                              float* __restrict__ means) {
  int idx = blockIdx.x * 256 + threadIdx.x;   // < 12288
  float s = 0.f;
  for (int m = 0; m < 128; ++m) s += part[(size_t)m * 12288 + idx];
  means[idx] = s * (1.0f / 8192.0f);
}

// ------------------------------------------------- top-2 SVD via subspace iter
// uv per matrix (256 floats): u_col0[64], u_col1[64], v_col0[64], v_col1[64]
__global__ void k_svd_top2(const float* __restrict__ means,
                           float* __restrict__ uv) {
  __shared__ float M[64][64];
  __shared__ float G[64][64];
  __shared__ float v0[64], v1[64], w0[64], w1[64], u0[64], u1[64];
  int t = threadIdx.x;                     // 64
  int mat = blockIdx.x;                    // 0:x3 1:x2 2:x1
  const float* src = means + mat * 4096;
  for (int k = t; k < 4096; k += 64) M[k >> 6][k & 63] = src[k];
  __syncthreads();
  for (int j = 0; j < 64; ++j) {           // G = M^T M (row t)
    float s = 0.f;
    for (int k = 0; k < 64; ++k) s += M[k][t] * M[k][j];
    G[t][j] = s;
  }
  v0[t] = 1.0f;
  v1[t] = (t & 1) ? 1.0f : -1.0f;
  __syncthreads();
  for (int it = 0; it < 32; ++it) {
    float a = 0.f, b = 0.f;
    for (int j = 0; j < 64; ++j) { a += G[t][j] * v0[j]; b += G[t][j] * v1[j]; }
    w0[t] = a; w1[t] = b;
    __syncthreads();
    if (t == 0) {                          // serial Gram-Schmidt (tiny)
      float n0 = 1e-20f;
      for (int j = 0; j < 64; ++j) n0 += w0[j] * w0[j];
      float r0 = rsqrtf(n0);
      for (int j = 0; j < 64; ++j) v0[j] = w0[j] * r0;
      float d = 0.f;
      for (int j = 0; j < 64; ++j) d += v0[j] * w1[j];
      float n1 = 1e-20f;
      for (int j = 0; j < 64; ++j) { w1[j] -= d * v0[j]; n1 += w1[j] * w1[j]; }
      float r1 = rsqrtf(n1);
      for (int j = 0; j < 64; ++j) v1[j] = w1[j] * r1;
    }
    __syncthreads();
  }
  {
    float a = 0.f, b = 0.f;
    for (int j = 0; j < 64; ++j) { a += M[t][j] * v0[j]; b += M[t][j] * v1[j]; }
    u0[t] = a; u1[t] = b;
  }
  __syncthreads();
  if (t == 0) {
    float n0 = 1e-20f, n1 = 1e-20f;
    for (int j = 0; j < 64; ++j) { n0 += u0[j] * u0[j]; n1 += u1[j] * u1[j]; }
    float r0 = rsqrtf(n0), r1 = rsqrtf(n1);
    for (int j = 0; j < 64; ++j) { u0[j] *= r0; u1[j] *= r1; }
  }
  __syncthreads();
  float* dst = uv + mat * 256;
  dst[t] = u0[t]; dst[64 + t] = u1[t]; dst[128 + t] = v0[t]; dst[192 + t] = v1[t];
}

// ------------------------------------------------- encode: s = u^T x v -> tokens
// scat[b, tok, c*4 + q*2 + r], tok = mat*64 + n  (s3|s2|s1 concat)
__global__ void k_encode(const float* __restrict__ x, const float* __restrict__ uv,
                         float* __restrict__ scat) {
  __shared__ float tile[4096];
  __shared__ float part[64][4][4];
  __shared__ float T3[64][2], T1[64][2];
  __shared__ float lv3[64][2], lv1[64][2], lu3[64][2], lu2[64][2], lu1[64][2];
  int t = threadIdx.x;                     // 256
  int bn = blockIdx.x;
  int b = bn >> 6, n = bn & 63;
  if (t < 64) {
    lu3[t][0] = uv[t];        lu3[t][1] = uv[64 + t];
    lv3[t][0] = uv[128 + t];  lv3[t][1] = uv[192 + t];
    lu2[t][0] = uv[256 + t];  lu2[t][1] = uv[320 + t];
    lu1[t][0] = uv[512 + t];  lu1[t][1] = uv[576 + t];
    lv1[t][0] = uv[640 + t];  lv1[t][1] = uv[704 + t];
  }
  __syncthreads();
  int i = t >> 2, g = t & 3;
  for (int c = 0; c < 64; ++c) {
    const float* tp = x + ((size_t)((b * 64 + c) * 64 + n)) * 4096;
#pragma unroll
    for (int k = 0; k < 4; ++k) {          // 16KB tile via async copy engine
      int vec = t + k * 256;
      async_copy_b128(lds_off(&tile[vec * 4]), tp + vec * 4);
    }
    wait_async();
    __syncthreads();
    float p0 = 0.f, p1 = 0.f, p2 = 0.f, p3 = 0.f;
    const float* row = tile + i * 64;
#pragma unroll
    for (int e = 0; e < 16; ++e) {
      int j = g * 16 + e;
      float xv = row[j];
      p0 += xv * lv3[j][0]; p1 += xv * lv3[j][1];
      p2 += xv * lv1[j][0]; p3 += xv * lv1[j][1];
    }
    part[i][g][0] = p0; part[i][g][1] = p1; part[i][g][2] = p2; part[i][g][3] = p3;
    __syncthreads();
    if (t < 64) {
      T3[t][0] = part[t][0][0] + part[t][1][0] + part[t][2][0] + part[t][3][0];
      T3[t][1] = part[t][0][1] + part[t][1][1] + part[t][2][1] + part[t][3][1];
      T1[t][0] = part[t][0][2] + part[t][1][2] + part[t][2][2] + part[t][3][2];
      T1[t][1] = part[t][0][3] + part[t][1][3] + part[t][2][3] + part[t][3][3];
    }
    __syncthreads();
    if (t < 12) {
      int mat = t >> 2, q = (t >> 1) & 1, r = t & 1;
      float s = 0.f;
      for (int k2 = 0; k2 < 64; ++k2) {
        float uval = (mat == 0) ? lu3[k2][q] : (mat == 1 ? lu2[k2][q] : lu1[k2][q]);
        float tval = (mat == 0) ? T3[k2][r] : T1[k2][r];
        s += uval * tval;
      }
      int tok = mat * 64 + n;
      scat[((size_t)(b * 192 + tok)) * 256 + c * 4 + q * 2 + r] = s;
    }
    __syncthreads();
  }
}

// ------------------------------------------------- layernorm -> bf16
__global__ void k_layernorm(const float* __restrict__ scat,
                            const float* __restrict__ gamma,
                            const float* __restrict__ beta,
                            unsigned short* __restrict__ xabf) {
  __shared__ float red[256];
  int tok = blockIdx.x, t = threadIdx.x;
  float v = scat[(size_t)tok * 256 + t];
  red[t] = v; __syncthreads();
  for (int s = 128; s > 0; s >>= 1) { if (t < s) red[t] += red[t + s]; __syncthreads(); }
  float mu = red[0] * (1.0f / 256.0f);
  __syncthreads();
  float d = v - mu;
  red[t] = d * d; __syncthreads();
  for (int s = 128; s > 0; s >>= 1) { if (t < s) red[t] += red[t + s]; __syncthreads(); }
  float var = red[0] * (1.0f / 256.0f);
  float xn = d * rsqrtf(var + 1e-5f) * gamma[t] + beta[t];
  xabf[(size_t)tok * 256 + t] = f2bf(xn);
}

// ------------------------------------------------- bf16 WMMA GEMM (batched)
// C[M,N] f32 = A[M,K]bf16(row-major) * Bt[N,K]bf16(row-major, i.e. B transposed)
// 128 threads = 4 waves, 32x32 macro-tile, one 16x16 WMMA per wave per k-step.
// LDS tiles are stored in WMMA *fragment-major* layout so every lane reads its
// whole fragment as 32 contiguous bytes (2x ds_load_b128), and every staging
// thread writes exactly one async b128 per tile.
template <int K>
__global__ void k_gemm_bt(const unsigned short* __restrict__ A,
                          const unsigned short* __restrict__ Bt,
                          float* __restrict__ C,
                          int lda, int ldbt, int ldc,
                          long aBatch, long bBatch, long cBatch) {
  __shared__ __attribute__((aligned(32))) unsigned short Af[2][32][16];
  __shared__ __attribute__((aligned(32))) unsigned short Bf[2][32][16];
  int tid = threadIdx.x;                   // 128
  int bz = blockIdx.z;
  const unsigned short* Ab = A + (size_t)bz * aBatch;
  const unsigned short* Bb = Bt + (size_t)bz * bBatch;
  float* Cb = C + (size_t)bz * cBatch;
  int m0 = blockIdx.y * 32;
  int n0 = blockIdx.x * 32;
  int w = tid >> 5;                        // wave 0..3
  int lane = tid & 31;                     // wave32
  int wm = (w >> 1);                       // warp-row 0/1 (16 rows each)
  int wn = (w & 1);                        // warp-col 0/1
  int half = lane >> 4;
  int l15 = lane & 15;

  // staging decomposition: thread owns (row r, 8-short chunk c4) of a 32x32 tile
  int r  = tid >> 2;                       // 0..31
  int c4 = (tid & 3) * 8;                  // 0,8,16,24
  // A fragment-major dest: lane half selects k in {c4&8}, elem8 selects k>=16
  int a_lane = ((c4 & 8) ? 16 : 0) + (r & 15);
  int a_e0   = (c4 & 16) ? 8 : 0;
  unsigned a_dst = lds_off(&Af[r >> 4][a_lane][a_e0]);
  // B fragment-major dest: lane half selects k in {c4&8}, elem8 selects k>=16
  int b_lane = ((c4 & 8) ? 16 : 0) + (r & 15);
  int b_e0   = (c4 & 16) ? 8 : 0;
  unsigned b_dst = lds_off(&Bf[r >> 4][b_lane][b_e0]);

  v8f acc = {};
#pragma unroll
  for (int k0 = 0; k0 < K; k0 += 32) {
    async_copy_b128(a_dst, &Ab[(size_t)(m0 + r) * lda  + k0 + c4]);
    async_copy_b128(b_dst, &Bb[(size_t)(n0 + r) * ldbt + k0 + c4]);
    if (k0 + 32 < K)                       // global_prefetch_b8 of next A tile
      __builtin_prefetch(&Ab[(size_t)(m0 + r) * lda + k0 + 32 + c4], 0, 1);
    wait_async();                          // each wave drains its own ASYNCcnt
    __syncthreads();                       // -> all tiles visible in LDS
    v16bf af = *(const v16bf*)&Af[wm][lane][0];   // 2x ds_load_b128
    v16bf bg = *(const v16bf*)&Bf[wn][lane][0];   // 2x ds_load_b128
    acc = __builtin_amdgcn_wmma_f32_16x16x32_bf16(false, af, false, bg,
                                                  (short)0, acc, false, false);
    __syncthreads();                       // protect LDS before next stage
  }
  // D layout: VGPR v -> M = v + half*8, lane l15 -> N
#pragma unroll
  for (int v = 0; v < 8; ++v) {
    int row = m0 + wm * 16 + half * 8 + v;
    int col = n0 + wn * 16 + l15;
    Cb[(size_t)row * ldc + col] = acc[v];
  }
}

// ------------------------------------------------- split qkv -> Q,K row / V^T bf16
__global__ void k_pack_qkv(const float* __restrict__ qkv,
                           unsigned short* __restrict__ Qb,
                           unsigned short* __restrict__ Kb,
                           unsigned short* __restrict__ Vt) {
  int bt = blockIdx.x;                     // b*192 + tok
  int b = bt / 192, tok = bt % 192;
  int t = threadIdx.x;
  const float* src = qkv + (size_t)bt * 1536;
#pragma unroll
  for (int k = 0; k < 6; ++k) {
    int e = t + k * 256;
    int which = e >> 9;
    int inner = e & 511;
    int h = inner >> 6, d = inner & 63;
    float v = src[e];
    if (which == 0)
      Qb[((size_t)((b * 8 + h) * 192 + tok)) * 64 + d] = f2bf(v * 0.125f);
    else if (which == 1)
      Kb[((size_t)((b * 8 + h) * 192 + tok)) * 64 + d] = f2bf(v);
    else                                   // V stored transposed: [bh][d][tok]
      Vt[((size_t)((b * 8 + h) * 64 + d)) * 192 + tok] = f2bf(v);
  }
}

// ------------------------------------------------- softmax rows (192) -> bf16 P
__global__ void k_softmax(const float* __restrict__ S,
                          unsigned short* __restrict__ P) {
  __shared__ float red[64];
  int row = blockIdx.x;                    // bh*192 + row
  int t = threadIdx.x;                     // 64
  const float* src = S + (size_t)row * 192;
  float a = src[t], b = src[t + 64], c = src[t + 128];
  float m = fmaxf(a, fmaxf(b, c));
  red[t] = m; __syncthreads();
  for (int s = 32; s > 0; s >>= 1) { if (t < s) red[t] = fmaxf(red[t], red[t + s]); __syncthreads(); }
  m = red[0]; __syncthreads();
  float ea = __expf(a - m), eb = __expf(b - m), ec = __expf(c - m);
  red[t] = ea + eb + ec; __syncthreads();
  for (int s = 32; s > 0; s >>= 1) { if (t < s) red[t] += red[t + s]; __syncthreads(); }
  float inv = 1.0f / red[0];
  unsigned short* dst = P + (size_t)row * 192;
  dst[t] = f2bf(ea * inv); dst[t + 64] = f2bf(eb * inv); dst[t + 128] = f2bf(ec * inv);
}

// ------------------------------------------------- heads -> [b*192, 512] bf16
__global__ void k_pack_o(const float* __restrict__ O,
                         unsigned short* __restrict__ Obf) {
  int bt = blockIdx.x;
  int b = bt / 192, tok = bt % 192;
  int t = threadIdx.x;
#pragma unroll
  for (int k = 0; k < 2; ++k) {
    int e = t + k * 256;                   // 0..511
    int h = e >> 6, d = e & 63;
    float v = O[((size_t)((b * 8 + h) * 192 + tok)) * 64 + d];
    Obf[(size_t)bt * 512 + e] = f2bf(v);
  }
}

// ------------------------------------------------- decode: y1+y2+y3 (with swaps)
__global__ void k_decode(const float* __restrict__ sout,
                         const float* __restrict__ uv,
                         float* __restrict__ out) {
  __shared__ float UV[768];
  __shared__ float S3[64][4], S2[64][4], S1[64][4];
  int bc = blockIdx.x;
  int b = bc >> 6, c = bc & 63;
  int t = threadIdx.x;                     // 256
  for (int k = t; k < 768; k += 256) UV[k] = uv[k];
  for (int k = t; k < 768; k += 256) {
    int tok = k >> 2, e = k & 3;
    float v = sout[((size_t)(b * 192 + tok)) * 256 + c * 4 + e];
    if (tok < 64)       S3[tok][e] = v;
    else if (tok < 128) S2[tok - 64][e] = v;
    else                S1[tok - 128][e] = v;
  }
  __syncthreads();
  const float* u3c0 = UV;        const float* u3c1 = UV + 64;
  const float* v3c0 = UV + 128;  const float* v3c1 = UV + 192;
  const float* u2c0 = UV + 256;  const float* u2c1 = UV + 320;
  const float* v2c0 = UV + 384;  const float* v2c1 = UV + 448;
  const float* u1c0 = UV + 512;  const float* u1c1 = UV + 576;
  const float* v1c0 = UV + 640;  const float* v1c1 = UV + 704;
  float* obase = out + (size_t)bc * 262144;
  for (int m = 0; m < 64; ++m) {
    float u2m0 = u2c0[m], u2m1 = u2c1[m];
    float v1m0 = v1c0[m], v1m1 = v1c1[m];
    float s3a = S3[m][0], s3b = S3[m][1], s3c = S3[m][2], s3d = S3[m][3];
    float* orow = obase + (size_t)m * 4096;
#pragma unroll 4
    for (int k = 0; k < 16; ++k) {
      int idx = t + k * 256;
      int i = idx >> 6, j = idx & 63;
      float a0 = s3a * v3c0[j] + s3b * v3c1[j];       // y3 = u3 S3[m] v3^T
      float a1 = s3c * v3c0[j] + s3d * v3c1[j];
      float val = u3c0[i] * a0 + u3c1[i] * a1;
      float b0 = S2[i][0] * v2c0[j] + S2[i][1] * v2c1[j];   // y2 (swap 2,3)
      float b1 = S2[i][2] * v2c0[j] + S2[i][3] * v2c1[j];
      val += u2m0 * b0 + u2m1 * b1;
      float c0 = S1[j][0] * v1m0 + S1[j][1] * v1m1;         // y1 (swap 2,4)
      float c1 = S1[j][2] * v1m0 + S1[j][3] * v1m1;
      val += u1c0[i] * c0 + u1c1[i] * c1;
      orow[idx] = val;
    }
  }
}

// ---------------------------------------------------------------------------
extern "C" void kernel_launch(void* const* d_in, const int* in_sizes, int n_in,
                              void* d_out, int out_size, void* d_ws, size_t ws_size,
                              hipStream_t stream) {
  (void)in_sizes; (void)n_in; (void)out_size; (void)ws_size;
  const float* x     = (const float*)d_in[0];
  const float* gamma = (const float*)d_in[1];
  const float* beta  = (const float*)d_in[2];
  const float* w_qkv = (const float*)d_in[3];
  const float* w_out = (const float*)d_in[4];
  float* out = (float*)d_out;

  char* ws = (char*)d_ws;
  size_t off = 0;
  auto alloc = [&](size_t bytes) -> char* {
    char* p = ws + off;
    off = (off + bytes + 255) & ~(size_t)255;
    return p;
  };
  float*          part   = (float*)alloc(128 * 12288 * sizeof(float));
  float*          means  = (float*)alloc(3 * 4096 * sizeof(float));
  float*          uv     = (float*)alloc(768 * sizeof(float));
  float*          scat   = (float*)alloc(384 * 256 * sizeof(float));
  unsigned short* xabf   = (unsigned short*)alloc(384 * 256 * 2);
  unsigned short* wqkvT  = (unsigned short*)alloc((size_t)1536 * 256 * 2);
  unsigned short* woutT  = (unsigned short*)alloc((size_t)256 * 512 * 2);
  float*          qkv    = (float*)alloc((size_t)384 * 1536 * sizeof(float));
  unsigned short* Qb     = (unsigned short*)alloc((size_t)16 * 192 * 64 * 2);
  unsigned short* Kb     = (unsigned short*)alloc((size_t)16 * 192 * 64 * 2);
  unsigned short* Vt     = (unsigned short*)alloc((size_t)16 * 64 * 192 * 2);
  float*          S      = (float*)alloc((size_t)16 * 192 * 192 * sizeof(float));
  unsigned short* P      = (unsigned short*)alloc((size_t)16 * 192 * 192 * 2);
  float*          O      = (float*)alloc((size_t)16 * 192 * 64 * sizeof(float));
  unsigned short* Obf    = (unsigned short*)alloc((size_t)384 * 512 * 2);
  float*          sout   = (float*)alloc((size_t)384 * 256 * sizeof(float));

  // one-time weight transposes (B operands consumed as Bt[N,K])
  k_cast_transpose<<<(256 * 1536 + 255) / 256, 256, 0, stream>>>(w_qkv, wqkvT, 256, 1536);
  k_cast_transpose<<<(512 * 256 + 255) / 256, 256, 0, stream>>>(w_out, woutT, 512, 256);
  k_means_partial<<<128, 256, 0, stream>>>(x, part);
  k_means_final<<<48, 256, 0, stream>>>(part, means);
  k_svd_top2<<<3, 64, 0, stream>>>(means, uv);
  k_encode<<<128, 256, 0, stream>>>(x, uv, scat);
  k_layernorm<<<384, 256, 0, stream>>>(scat, gamma, beta, xabf);
  // qkv = xa @ w_qkv : M=384 N=1536 K=256
  k_gemm_bt<256><<<dim3(48, 12, 1), 128, 0, stream>>>(xabf, wqkvT, qkv,
      256, 256, 1536, 0L, 0L, 0L);
  k_pack_qkv<<<384, 256, 0, stream>>>(qkv, Qb, Kb, Vt);
  // S = Q @ K^T per (b,h): M=192 N=192 K=64 (Bt == K row-major), batch=16
  k_gemm_bt<64><<<dim3(6, 6, 16), 128, 0, stream>>>(Qb, Kb, S,
      64, 64, 192, (long)192 * 64, (long)192 * 64, (long)192 * 192);
  k_softmax<<<3072, 64, 0, stream>>>(S, P);
  // O = P @ V : M=192 N=64 K=192 (Bt == V^T [64,192]), batch=16
  k_gemm_bt<192><<<dim3(2, 6, 16), 128, 0, stream>>>(P, Vt, O,
      192, 192, 64, (long)192 * 192, (long)64 * 192, (long)192 * 64);
  k_pack_o<<<384, 256, 0, stream>>>(O, Obf);
  // s_out = O @ w_out : M=384 N=256 K=512
  k_gemm_bt<512><<<dim3(8, 12, 1), 128, 0, stream>>>(Obf, woutT, sout,
      512, 512, 256, 0L, 0L, 0L);
  k_decode<<<128, 256, 0, stream>>>(sout, uv, out);
}
